// CapsuleBlock_7997229105204
// MI455X (gfx1250) — matrix operand order repaired and checked by reading, not compiled, not run on hardware
//
#include <hip/hip_runtime.h>
#include <hip/hip_bf16.h>

typedef __bf16 bf16_t;
typedef __bf16 v16bf __attribute__((ext_vector_type(16)));
typedef float  v8f   __attribute__((ext_vector_type(8)));
typedef unsigned int v4u __attribute__((ext_vector_type(4)));
typedef int v8i __attribute__((ext_vector_type(8)));
typedef int v4i __attribute__((ext_vector_type(4)));

#define NB   32      // batch
#define NC   32      // output capsules
#define NI   4096    // input capsules
#define ND   16      // capsule dim
#define EPSF 1e-7f

#define WAVES_PER_BH  256
#define BLOCKS_PER_BH 128          // 2 waves per block
#define I_PER_WAVE    (NI / WAVES_PER_BH)   // 16

// LDS float-offset map (dynamic shared, no static LDS in route kernel)
// W tiles are stored K-padded by the TDM: each 32 B j-row of data is followed
// by a 32 B always-zero hole => one tile = 32 c * 16 j * 64 B = 32 KB.
#define L_VSUM    0         // [16 b][32 c][16 j]            = 8192 f
#define L_WAVE    8192      // per-wave region base
#define L_WSTRIDE 32768     // floats per wave region
#define LW_UHAT   0         // [32 c][16 b][16 j]            = 8192 f
#define LW_VACC   8192      //                               = 8192 f
#define LW_WT0    16384     // padded W tile buf 0           = 8192 f (32 KB)
#define LW_WT1    24576     // padded W tile buf 1           = 8192 f
#define L_TOTAL_BYTES ((L_WAVE + 2 * L_WSTRIDE) * 4)   // 294912 B

__device__ __forceinline__ unsigned pack2bf(float a, float b) {
  union { bf16_t h[2]; unsigned u; } p;
  p.h[0] = (bf16_t)a; p.h[1] = (bf16_t)b;
  return p.u;
}

// ---------------- zero helper ----------------
__global__ void zero_f32(float* __restrict__ p, int n) {
  int t = blockIdx.x * blockDim.x + threadIdx.x;
  if (t < n) p[t] = 0.0f;
}

// ---------------- squash(x) -> u (bf16, layout [i][b][d]) ----------------
__global__ void prep_u(const float* __restrict__ x, bf16_t* __restrict__ ub) {
  int tid = blockIdx.x * blockDim.x + threadIdx.x;       // 32*4096 threads
  if (tid >= NB * NI) return;
  int b = tid >> 12;              // / NI
  int i = tid & (NI - 1);
  const float4* xr = (const float4*)(x + ((size_t)b * NI + i) * ND);
  float4 f0 = xr[0], f1 = xr[1], f2 = xr[2], f3 = xr[3];
  float v[16] = {f0.x,f0.y,f0.z,f0.w, f1.x,f1.y,f1.z,f1.w,
                 f2.x,f2.y,f2.z,f2.w, f3.x,f3.y,f3.z,f3.w};
  float sq = 0.0f;
#pragma unroll
  for (int k = 0; k < 16; ++k) sq += v[k] * v[k];
  float scale = sq / (1.0f + sq) * rsqrtf(sq + EPSF);
  uint4 o0, o1;
  o0.x = pack2bf(scale*v[0],  scale*v[1]);  o0.y = pack2bf(scale*v[2],  scale*v[3]);
  o0.z = pack2bf(scale*v[4],  scale*v[5]);  o0.w = pack2bf(scale*v[6],  scale*v[7]);
  o1.x = pack2bf(scale*v[8],  scale*v[9]);  o1.y = pack2bf(scale*v[10], scale*v[11]);
  o1.z = pack2bf(scale*v[12], scale*v[13]); o1.w = pack2bf(scale*v[14], scale*v[15]);
  uint4* dst = (uint4*)(ub + ((size_t)i * NB + b) * ND);
  dst[0] = o0; dst[1] = o1;
}

// ---------------- W f32 -> bf16 (same layout [c][i][j][d]) ----------------
__global__ void wconv(const float* __restrict__ w, bf16_t* __restrict__ wb) {
  long long t = (long long)blockIdx.x * blockDim.x + threadIdx.x;
  long long base = t * 8;
  if (base >= (long long)NC * NI * ND * ND) return;
  const float4* s = (const float4*)(w + base);
  float4 a = s[0], b = s[1];
  uint4 o;
  o.x = pack2bf(a.x, a.y); o.y = pack2bf(a.z, a.w);
  o.z = pack2bf(b.x, b.y); o.w = pack2bf(b.z, b.w);
  *(uint4*)(wb + base) = o;
}

// ---------------- TDM: load one [32 c][256] bf16 W tile to LDS -------------
// D# group0: count=1 | lds_addr | global_addr(57b) | type=2
// D# group1: data_size=2B; pad_enable, pad_interval=8 DW (32 B),
//            pad_amount=8 DW (32 B) -> K=32-padded rows in LDS;
//            tensor_dim0=256; tensor_dim1=32; tile_dim0=256; tile_dim1=32;
//            tensor_dim0_stride = NI*256 (data_size units)
__device__ __forceinline__ void tdm_load_wtile(unsigned long long gaddr,
                                               unsigned ldsoff) {
  unsigned g_lo = __builtin_amdgcn_readfirstlane((unsigned)(gaddr & 0xFFFFFFFFu));
  unsigned g_hi = __builtin_amdgcn_readfirstlane((unsigned)(gaddr >> 32));
  unsigned loff = __builtin_amdgcn_readfirstlane(ldsoff);
  v4u g0 = { 1u,                                   // count=1, user mode
             loff,                                 // lds_addr (bytes)
             g_lo,                                 // global_addr[31:0]
             (g_hi & 0x1FFFFFFu) | (2u << 30) };   // addr[56:32] | type=2
  v8i g1 = { (int)0x0E910000u,                     // dsize=2B|pad|int=8dw|amt=8dw
             (int)(256u << 16),                    // tensor_dim0 = 256
             (int)(32u << 16),                     // tensor_dim1 = 32
             (int)(256u << 16),                    // tile_dim0 = 256
             32,                                   // tile_dim1 = 32
             0x00100000,                           // dim0_stride = 1048576
             0, 0 };
  v4i gz4 = { 0, 0, 0, 0 };
  v8i gz8 = { 0, 0, 0, 0, 0, 0, 0, 0 };
  __builtin_amdgcn_tensor_load_to_lds(g0, g1, gz4, gz4, gz8, 0);
}

// ---------------- fused routing iteration ----------------
// grid: 2*BLOCKS_PER_BH blocks, 64 threads (2 waves).
// Per wave: TDM double-buffers K-padded W tiles into LDS, recomputes u_hat
// tiles (WMMA bf16, K 16->32 zero-padded), logits = vsum . u_hat, softmax
// over c, accumulates coupling-weighted u_hat into a per-wave partial.
__global__ void __launch_bounds__(64)
route_iter(const bf16_t* __restrict__ ub, const bf16_t* __restrict__ wb,
           const float* __restrict__ vsum_g, float* __restrict__ pv) {
  extern __shared__ float smem[];
  const int lane = threadIdx.x & 31;
  const int wave = threadIdx.x >> 5;          // 0..1
  const int bh   = blockIdx.x & 1;            // batch half
  const int blkI = blockIdx.x >> 1;           // 0..127
  const int wr   = blkI * 2 + wave;           // 0..255 wave rank within bh

  bf16_t* smem_b = (bf16_t*)smem;             // LDS base viewed as bf16
  const unsigned wbase_f = L_WAVE + wave * L_WSTRIDE;   // float units
  float*  vsum_l = smem + L_VSUM;
  float*  uhat_l = smem + wbase_f + LW_UHAT;
  float*  vacc_l = smem + wbase_f + LW_VACC;
  // bf16-element offsets of the two padded W tile buffers
  const unsigned wtu[2] = { (wbase_f + LW_WT0) * 2, (wbase_f + LW_WT1) * 2 };
  // LDS byte addresses for the TDM descriptors
  const unsigned wtbytes[2] = { (wbase_f + LW_WT0) * 4, (wbase_f + LW_WT1) * 4 };

  // cooperative load of this batch-half's vsum slice
  const float* vsrc = vsum_g + (size_t)bh * 16 * NC * ND;
  for (int k = threadIdx.x * 4; k < 8192; k += 64 * 4)
    *(float4*)(vsum_l + k) = *(const float4*)(vsrc + k);
  // zero per-wave accumulator AND both W tile buffers (the TDM pad holes in
  // the tile buffers must read as zero; data halves get overwritten by DMA)
  float4 f4z = {0.f, 0.f, 0.f, 0.f};
  for (int k = lane * 4; k < LW_WT1 + 8192 - LW_VACC; k += 128)
    *(float4*)(vacc_l + k) = f4z;
  __syncthreads();   // also orders the LDS zeroing before the first TDM write

  const int l15 = lane & 15;    // A-row (batch) / D-column (j) / owned-b index
  const int l16 = lane >> 4;    // K-half select / c-half select
  const uint4 zu = {0u, 0u, 0u, 0u};
  v8f czero = {0.f, 0.f, 0.f, 0.f, 0.f, 0.f, 0.f, 0.f};

  // fixed per-lane B-operand base inside a padded tile:
  // lane j+16h: elems 0-7 = data d 8h..8h+7 (K = 8h+s), elems 8-15 = zero hole
  const unsigned blane = (unsigned)l15 * 32u + (unsigned)l16 * 8u;  // bf16 elems

  const unsigned long long wb_base = (unsigned long long)(size_t)wb;
  const int i0 = wr * I_PER_WAVE;

  // prologue: start DMA of first W tile
  tdm_load_wtile(wb_base + (unsigned long long)i0 * (ND * ND) * 2, wtbytes[0]);

  for (int t = 0; t < I_PER_WAVE; ++t) {
    const int i = i0 + t;
    const int cur = t & 1;

    // kick off next tile's DMA, then wait for current tile
    if (t + 1 < I_PER_WAVE) {
      tdm_load_wtile(wb_base + (unsigned long long)(i + 1) * (ND * ND) * 2,
                     wtbytes[1 - cur]);
      __builtin_amdgcn_s_wait_tensorcnt((short)1);
    } else {
      __builtin_amdgcn_s_wait_tensorcnt((short)0);
    }
    __builtin_amdgcn_sched_barrier(0);
    asm volatile("" ::: "memory");

    // ---- A operand: u tile, M=batch(16), K=d(16 padded to 32) ----
    union { v16bf v; uint4 q[2]; } A;
    A.q[0] = *(const uint4*)(ub + (((size_t)i * NB) + bh * 16 + l15) * ND + l16 * 8);
    A.q[1] = zu;   // K = 16..31 zero pad

    // single base VGPR; per-c offsets are compile-time immediates
    const bf16_t* bp = smem_b + (wtu[cur] + blane);

    // ---- per output capsule: WMMA u_hat tile -> LDS ----
#pragma unroll
    for (int c = 0; c < NC; ++c) {
      union { v16bf v; uint4 q[2]; } Bm;
      Bm.q[0] = *(const uint4*)(bp + c * 512);        // ds offset = c*1024 B
      Bm.q[1] = *(const uint4*)(bp + c * 512 + 16);   //            + 32 B
      v8f d = __builtin_amdgcn_wmma_f32_16x16x32_bf16(
          false, A.v, false, Bm.v, (short)0, czero, false, false);
#pragma unroll
      for (int r = 0; r < 8; ++r) {
        int m = r + 8 * l16;   // batch row
        uhat_l[(c * 16 + m) * 16 + l15] = d[r];  // l15 here = column j
      }
    }

    // ---- logits: b[b,c,i] = sum_j vsum[b,c,j] * u_hat[b,c,i,j] ----
    float lg[16];
#pragma unroll
    for (int k = 0; k < 16; ++k) {
      int c = l16 * 16 + k;
      const float* up = uhat_l + (c * 16 + l15) * 16;
      const float* vp = vsum_l + (l15 * NC + c) * ND;
      float acc = 0.0f;
#pragma unroll
      for (int jq = 0; jq < 4; ++jq) {
        float4 u4 = *(const float4*)(up + jq * 4);
        float4 v4 = *(const float4*)(vp + jq * 4);
        acc += u4.x * v4.x + u4.y * v4.y + u4.z * v4.z + u4.w * v4.w;
      }
      lg[k] = acc;
    }
    float full[32];
#pragma unroll
    for (int k = 0; k < 16; ++k) {
      float o = __shfl_xor(lg[k], 16, 32);
      full[k]      = (l16 == 0) ? lg[k] : o;
      full[16 + k] = (l16 == 0) ? o : lg[k];
    }
    // softmax over c
    float mx = full[0];
#pragma unroll
    for (int k = 1; k < 32; ++k) mx = fmaxf(mx, full[k]);
    float cp[32];
    float s = 0.0f;
#pragma unroll
    for (int k = 0; k < 32; ++k) { float e = __expf(full[k] - mx); cp[k] = e; s += e; }
    float inv = 1.0f / s;

    // ---- accumulate coupling-weighted u_hat into per-wave v partial ----
#pragma unroll
    for (int k = 0; k < 16; ++k) {
      int c = l16 * 16 + k;
      float cc = cp[c] * inv;
      const float* up = uhat_l + (c * 16 + l15) * 16;
      float*       vp = vacc_l + (c * 16 + l15) * 16;
#pragma unroll
      for (int jq = 0; jq < 4; ++jq) {
        float4 u4 = *(const float4*)(up + jq * 4);
        float4 a4 = *(float4*)(vp + jq * 4);
        a4.x += cc * u4.x; a4.y += cc * u4.y; a4.z += cc * u4.z; a4.w += cc * u4.w;
        *(float4*)(vp + jq * 4) = a4;
      }
    }
  }

  // deterministic partial write-out (fixed per-wave slot, reduced later)
  float* dst = pv + ((size_t)bh * WAVES_PER_BH + wr) * 8192;
  for (int k = lane * 4; k < 8192; k += 128)
    *(float4*)(dst + k) = *(const float4*)(vacc_l + k);
}

// ---------------- fixed-order reduction of wave partials ----------------
__global__ void reduce_vacc(const float* __restrict__ pv, float* __restrict__ vacc) {
  int v = blockIdx.x * blockDim.x + threadIdx.x;   // 16384 = [b][c][j]
  if (v >= NB * NC * ND) return;
  int b = v >> 9;
  int c = (v >> 4) & 31;
  int j = v & 15;
  int bh = b >> 4, bl = b & 15;
  const float* src = pv + (size_t)bh * WAVES_PER_BH * 8192 + (c * 16 + bl) * 16 + j;
  float s = 0.0f;
  for (int w = 0; w < WAVES_PER_BH; ++w) s += src[(size_t)w * 8192];
  vacc[v] = s;
}

// ---------------- v = squash(vacc); vsum += v (or final write) ----------------
__global__ void squash_v(const float* __restrict__ vacc, float* __restrict__ vsum,
                         float* __restrict__ out, int is_final) {
  int bc = blockIdx.x * blockDim.x + threadIdx.x;  // 1024 = [b][c]
  if (bc >= NB * NC) return;
  const float4* p = (const float4*)(vacc + (size_t)bc * ND);
  float4 a = p[0], b = p[1], c = p[2], d = p[3];
  float sq = a.x*a.x + a.y*a.y + a.z*a.z + a.w*a.w
           + b.x*b.x + b.y*b.y + b.z*b.z + b.w*b.w
           + c.x*c.x + c.y*c.y + c.z*c.z + c.w*c.w
           + d.x*d.x + d.y*d.y + d.z*d.z + d.w*d.w;
  float scale = sq / (1.0f + sq) * rsqrtf(sq + EPSF);
  a.x*=scale; a.y*=scale; a.z*=scale; a.w*=scale;
  b.x*=scale; b.y*=scale; b.z*=scale; b.w*=scale;
  c.x*=scale; c.y*=scale; c.z*=scale; c.w*=scale;
  d.x*=scale; d.y*=scale; d.z*=scale; d.w*=scale;
  if (is_final) {
    float4* o = (float4*)(out + (size_t)bc * ND);
    o[0] = a; o[1] = b; o[2] = c; o[3] = d;
  } else {
    float4* vs = (float4*)(vsum + (size_t)bc * ND);
    float4 s0 = vs[0], s1 = vs[1], s2 = vs[2], s3 = vs[3];
    s0.x+=a.x; s0.y+=a.y; s0.z+=a.z; s0.w+=a.w;
    s1.x+=b.x; s1.y+=b.y; s1.z+=b.z; s1.w+=b.w;
    s2.x+=c.x; s2.y+=c.y; s2.z+=c.z; s2.w+=c.w;
    s3.x+=d.x; s3.y+=d.y; s3.z+=d.z; s3.w+=d.w;
    vs[0]=s0; vs[1]=s1; vs[2]=s2; vs[3]=s3;
  }
}

extern "C" void kernel_launch(void* const* d_in, const int* in_sizes, int n_in,
                              void* d_out, int out_size, void* d_ws, size_t ws_size,
                              hipStream_t stream) {
  const float* x = (const float*)d_in[0];   // [32, 65536]
  const float* W = (const float*)d_in[1];   // [32, 4096, 16, 16]
  float* out = (float*)d_out;               // [32, 32, 16]

  char* ws = (char*)d_ws;
  bf16_t* ub   = (bf16_t*)(ws);                                   //  4 MB
  bf16_t* wb   = (bf16_t*)(ws + (4ll  << 20));                    // 64 MB
  float*  pv   = (float*) (ws + (68ll << 20));                    // 16 MB
  float*  vacc = (float*) (ws + (84ll << 20));                    // 64 KB
  float*  vsum = (float*) (ws + (84ll << 20) + (64ll << 10));     // 64 KB

  // reset iteration state (b_logits implicit via vsum)
  zero_f32<<<(NB*NC*ND + 255)/256, 256, 0, stream>>>(vsum, NB*NC*ND);
  // squash input capsules -> bf16, WMMA-friendly [i][b][d] layout
  prep_u<<<(NB*NI + 255)/256, 256, 0, stream>>>(x, ub);
  // W -> bf16 (L2-resident working set for routing passes)
  wconv<<<(int)(((long long)NC*NI*ND*ND/8 + 255)/256), 256, 0, stream>>>(W, wb);

  for (int t = 0; t < 3; ++t) {
    route_iter<<<2*BLOCKS_PER_BH, 64, L_TOTAL_BYTES, stream>>>(ub, wb, vsum, pv);
    reduce_vacc<<<(NB*NC*ND + 255)/256, 256, 0, stream>>>(pv, vacc);
    squash_v<<<(NB*NC + 255)/256, 256, 0, stream>>>(vacc, vsum, out, t == 2 ? 1 : 0);
  }
}
